// GMMILDiscriminator_15882789060993
// MI455X (gfx1250) — matrix operand order repaired and checked by reading, not compiled, not run on hardware
//
#include <hip/hip_runtime.h>
#include <hip/hip_bf16.h>

// ---------------------------------------------------------------------------
// GMMIL discriminator reward (state_only) for MI455X / gfx1250.
//
// D_es = ||e_i||^2/d + ||s_j||^2/d - 2 e_i.s_j/d      (4096x4096, d=512)
// D_cc = col-gram of expert_state (512x512, d=4096)    -> gamma_2
// gamma_1 = 1/median(D_es), gamma_2 = 1/median(D_cc)
// out[j] = mean_i[exp(-g1 D_es)+exp(-g2 D_es)] - mean_i[exp(-g1 D_ss)+exp(-g2 D_ss)]
//
// GEMMs run on v_wmma_f32_16x16x32_bf16 with split-bf16 (hi+lo) inputs:
// dot ~= hi.hi + hi.lo + lo.hi  (3 WMMAs per 32-deep K chunk, ~f32 accuracy).
// K-loop is hoisted outside the row-tile loop: B fragments are loaded once per
// K-chunk and reused across T row tiles held in T independent accumulators
// (breaks WMMA->WMMA RAW chains, cuts B vmem traffic by T).
// Medians via deterministic integer histograms (LDS-staged).
// ---------------------------------------------------------------------------

typedef __attribute__((ext_vector_type(16))) __bf16 bf16x16;
typedef __attribute__((ext_vector_type(8)))  __bf16 bf16x8;
typedef __attribute__((ext_vector_type(8)))  float  f32x8;

#define NBINS     8192
#define BIN_SCALE 1024.0f          // range [0, 8)
#define BIN_W     (1.0f / 1024.0f)

// ---------------- fragment loads (contiguous k order, shared by A and B) -------------

// A fragment 16x32 bf16: lane L<16 -> row m0+L, K = 0..15;
//                        lane L+16 -> same row, K = 16..31.
__device__ __forceinline__ bf16x16 load_a_frag(const __bf16* X, int ld,
                                               int m0, int kc, int lane) {
  const int half = lane >> 4;
  const __bf16* p = X + (size_t)(m0 + (lane & 15)) * (size_t)ld + kc + half * 16;
  bf16x8 x = *(const bf16x8*)(p);
  bf16x8 y = *(const bf16x8*)(p + 8);
  bf16x16 r;
#pragma unroll
  for (int i = 0; i < 8; ++i) { r[i] = x[i]; r[i + 8] = y[i]; }
  return r;
}

// B fragment 32x16 bf16 (B = Y^T, Y row-major [N][K]):
// lane L<16 -> col n0+L, K = kc..kc+15 ; lane L+16 -> same col, K = kc+16..kc+31.
__device__ __forceinline__ bf16x16 load_b_frag(const __bf16* Y, int ld,
                                               int n0, int kc, int lane) {
  const int half = lane >> 4;
  const __bf16* p = Y + (size_t)(n0 + (lane & 15)) * (size_t)ld + kc + half * 16;
  return *(const bf16x16*)p;
}

// T stacked 16x16 tiles (rows mbase + t*16, cols n0..n0+15) of X.Y^T.
// K-chunk outer loop: B hi/lo fragments loaded once and reused across the
// T row tiles; T independent accumulator chains feed the XDL pipe.
template <int T>
__device__ __forceinline__ void panel_dot(f32x8 (&c)[T],
                                          const __bf16* __restrict__ Xh,
                                          const __bf16* __restrict__ Xl,
                                          const __bf16* __restrict__ Yh,
                                          const __bf16* __restrict__ Yl,
                                          int ld, int K, int mbase, int n0, int lane) {
#pragma unroll
  for (int t = 0; t < T; ++t)
    c[t] = (f32x8){0.f, 0.f, 0.f, 0.f, 0.f, 0.f, 0.f, 0.f};
  for (int kc = 0; kc < K; kc += 32) {
    const bf16x16 bh = load_b_frag(Yh, ld, n0, kc, lane);
    const bf16x16 bl = load_b_frag(Yl, ld, n0, kc, lane);
#pragma unroll
    for (int t = 0; t < T; ++t) {
      const bf16x16 ah = load_a_frag(Xh, ld, mbase + t * 16, kc, lane);
      const bf16x16 al = load_a_frag(Xl, ld, mbase + t * 16, kc, lane);
      c[t] = __builtin_amdgcn_wmma_f32_16x16x32_bf16(false, ah, false, bh, (short)0, c[t], false, false);
      c[t] = __builtin_amdgcn_wmma_f32_16x16x32_bf16(false, ah, false, bl, (short)0, c[t], false, false);
      c[t] = __builtin_amdgcn_wmma_f32_16x16x32_bf16(false, al, false, bh, (short)0, c[t], false, false);
    }
  }
}

// ---------------- prep kernels ---------------------------------------------

__global__ void zero_u32_kernel(unsigned int* p, int n) {
  int i = blockIdx.x * blockDim.x + threadIdx.x;
  if (i < n) p[i] = 0u;
}

// split f32 -> bf16 hi + bf16 residual
__global__ void split_kernel(const float* __restrict__ X, __bf16* __restrict__ H,
                             __bf16* __restrict__ L, int n) {
  int i = blockIdx.x * blockDim.x + threadIdx.x;
  if (i >= n) return;
  float v = X[i];
  __bf16 h = (__bf16)v;
  float r = v - (float)h;
  H[i] = h;
  L[i] = (__bf16)r;
}

// split + transposed copy (for the column-gram, keeps WMMA loads contiguous)
__global__ void split_t_kernel(const float* __restrict__ X,
                               __bf16* __restrict__ H, __bf16* __restrict__ L,
                               __bf16* __restrict__ Ht, __bf16* __restrict__ Lt,
                               int rows, int cols) {
  int i = blockIdx.x * blockDim.x + threadIdx.x;
  if (i >= rows * cols) return;
  int r = i / cols, c = i - r * cols;
  float v = X[i];
  __bf16 h = (__bf16)v;
  float res = v - (float)h;
  __bf16 l = (__bf16)res;
  H[i] = h; L[i] = l;
  Ht[(size_t)c * rows + r] = h;
  Lt[(size_t)c * rows + r] = l;
}

__global__ void rownorm_kernel(const float* __restrict__ X, int d,
                               float* __restrict__ out) {
  const int row = blockIdx.x;
  const float* p = X + (size_t)row * d;
  float s = 0.f;
  for (int k = threadIdx.x; k < d; k += blockDim.x) { float v = p[k]; s += v * v; }
  __shared__ float sm[256];
  sm[threadIdx.x] = s; __syncthreads();
  for (int o = 128; o > 0; o >>= 1) {
    if (threadIdx.x < o) sm[threadIdx.x] += sm[threadIdx.x + o];
    __syncthreads();
  }
  if (threadIdx.x == 0) out[row] = sm[0];
}

__global__ void colnorm_kernel(const float* __restrict__ X, int rows, int cols,
                               float* __restrict__ out) {
  const int col = blockIdx.x;
  float s = 0.f;
  for (int r = threadIdx.x; r < rows; r += blockDim.x) {
    float v = X[(size_t)r * cols + col];
    s += v * v;
  }
  __shared__ float sm[256];
  sm[threadIdx.x] = s; __syncthreads();
  for (int o = 128; o > 0; o >>= 1) {
    if (threadIdx.x < o) sm[threadIdx.x] += sm[threadIdx.x + o];
    __syncthreads();
  }
  if (threadIdx.x == 0) out[col] = sm[0];
}

// ---------------- histogram of squared-distance matrix ----------------------
// Block = 256 threads (8 waves). Block covers 256 rows x 16 cols: wave w
// computes 2 stacked 16x16 tiles. Per-block LDS histogram, integer flush.

__global__ void __launch_bounds__(256)
dist_hist_kernel(const __bf16* __restrict__ Xh, const __bf16* __restrict__ Xl,
                 const float* __restrict__ x2,
                 const __bf16* __restrict__ Yh, const __bf16* __restrict__ Yl,
                 const float* __restrict__ y2,
                 int ld, int K, float invd, unsigned int* __restrict__ hist) {
  __shared__ unsigned int lh[NBINS];
  for (int i = threadIdx.x; i < NBINS; i += 256) lh[i] = 0u;
  __syncthreads();

  const int lane = threadIdx.x & 31;
  const int wave = threadIdx.x >> 5;
  const int n0 = blockIdx.x * 16;
  const int mbase = blockIdx.y * 256 + wave * 32;
  const int half = lane >> 4;
  const float yn = y2[n0 + (lane & 15)];

  f32x8 c[2];
  panel_dot<2>(c, Xh, Xl, Yh, Yl, ld, K, mbase, n0, lane);

#pragma unroll
  for (int t = 0; t < 2; ++t) {
#pragma unroll
    for (int r = 0; r < 8; ++r) {
      const int m = mbase + t * 16 + r + half * 8;
      const float D = (x2[m] + yn - 2.0f * c[t][r]) * invd;
      int b = (int)(D * BIN_SCALE);
      b = b < 0 ? 0 : (b >= NBINS ? NBINS - 1 : b);
      atomicAdd(&lh[b], 1u);
    }
  }
  __syncthreads();
  for (int i = threadIdx.x; i < NBINS; i += 256) {
    unsigned int v = lh[i];
    if (v) atomicAdd(&hist[i], v);
  }
}

// ---------------- median -> gamma ------------------------------------------

__global__ void median_gamma_kernel(const unsigned int* __restrict__ h1, unsigned int n1,
                                    const unsigned int* __restrict__ h2, unsigned int n2,
                                    float* __restrict__ g) {
  const int t = threadIdx.x;
  if (t > 1) return;
  const unsigned int* h = (t == 0) ? h1 : h2;
  const unsigned int total = (t == 0) ? n1 : n2;
  const unsigned int target = (total + 1u) >> 1;
  unsigned int cum = 0, cnt = 0;
  int b = 0;
  for (b = 0; b < NBINS; ++b) {
    cnt = h[b];
    if (cum + cnt >= target) break;
    cum += cnt;
  }
  if (b >= NBINS) { b = NBINS - 1; cnt = 1u; }
  float frac = cnt ? ((float)(target - cum)) / (float)cnt : 0.5f;
  float med = ((float)b + frac) * BIN_W;
  g[t] = 1.0f / med;
}

// ---------------- exp-kernel sums (deterministic partials) ------------------
// grid = (N/16, 4); block = 256 (8 waves). Wave w owns columns [16*bx,16*bx+16)
// and row chunk (by*8+w)*128 .. +128 held as 8 accumulator tiles.
// Writes partial[chunk][col] -> fixed-order reduction in finalize (no float
// atomics anywhere => bitwise deterministic across graph replays).

__global__ void __launch_bounds__(256)
expsum_kernel(const __bf16* __restrict__ Xh, const __bf16* __restrict__ Xl,
              const float* __restrict__ x2,
              const __bf16* __restrict__ Yh, const __bf16* __restrict__ Yl,
              const float* __restrict__ y2,
              int ld, int K, float invd, int N,
              const float* __restrict__ g, float* __restrict__ partial) {
  const int lane = threadIdx.x & 31;
  const int wave = threadIdx.x >> 5;
  const int n0 = blockIdx.x * 16;
  const int chunk = blockIdx.y * 8 + wave;   // 0..31
  const int mbase = chunk * 128;
  const int half = lane >> 4;
  const float g1 = g[0], g2 = g[1];
  const float yn = y2[n0 + (lane & 15)];

  f32x8 c[8];
  panel_dot<8>(c, Xh, Xl, Yh, Yl, ld, K, mbase, n0, lane);

  float acc = 0.f;
#pragma unroll
  for (int t = 0; t < 8; ++t) {
#pragma unroll
    for (int r = 0; r < 8; ++r) {
      const int m = mbase + t * 16 + r + half * 8;
      const float D = (x2[m] + yn - 2.0f * c[t][r]) * invd;
      acc += __expf(-g1 * D) + __expf(-g2 * D);
    }
  }
  __shared__ float tmp[256];
  tmp[threadIdx.x] = acc;
  __syncthreads();
  if (lane < 16)
    partial[(size_t)chunk * N + n0 + lane] = tmp[threadIdx.x] + tmp[threadIdx.x + 16];
}

__global__ void finalize_kernel(const float* __restrict__ Pes,
                                const float* __restrict__ Pss,
                                float* __restrict__ out, int N) {
  int j = blockIdx.x * blockDim.x + threadIdx.x;
  if (j >= N) return;
  float a = 0.f, b = 0.f;
  for (int c = 0; c < 32; ++c) {
    a += Pes[(size_t)c * N + j];
    b += Pss[(size_t)c * N + j];
  }
  out[j] = (a - b) * (1.0f / 4096.0f);
}

// ---------------------------------------------------------------------------

extern "C" void kernel_launch(void* const* d_in, const int* in_sizes, int n_in,
                              void* d_out, int out_size, void* d_ws, size_t ws_size,
                              hipStream_t stream) {
  (void)in_sizes; (void)n_in; (void)out_size; (void)ws_size;
  const float* state  = (const float*)d_in[0];   // [4096,512]
  const float* expert = (const float*)d_in[2];   // [4096,512]  (actions unused)
  float* out = (float*)d_out;

  const int N = 4096, D = 512;
  const size_t ND = (size_t)N * D;

  // workspace carve-up (256B aligned)
  char* ws = (char*)d_ws;
  size_t off = 0;
  auto alloc = [&](size_t bytes) -> char* {
    char* p = ws + off;
    off += (bytes + 255) & ~(size_t)255;
    return p;
  };
  __bf16* Sh  = (__bf16*)alloc(ND * 2);
  __bf16* Sl  = (__bf16*)alloc(ND * 2);
  __bf16* Eh  = (__bf16*)alloc(ND * 2);
  __bf16* El  = (__bf16*)alloc(ND * 2);
  __bf16* Eth = (__bf16*)alloc(ND * 2);          // expert transposed [512][4096]
  __bf16* Etl = (__bf16*)alloc(ND * 2);
  float*  s2  = (float*)alloc(N * 4);
  float*  e2  = (float*)alloc(N * 4);
  float*  c2  = (float*)alloc(D * 4);
  unsigned int* hist1 = (unsigned int*)alloc(NBINS * 4);
  unsigned int* hist2 = (unsigned int*)alloc(NBINS * 4);
  float*  gam = (float*)alloc(256);
  float*  Pes = (float*)alloc((size_t)32 * N * 4);
  float*  Pss = (float*)alloc((size_t)32 * N * 4);

  // 1) zero histograms (harness does not re-poison between replays)
  zero_u32_kernel<<<dim3((2 * NBINS + 255) / 256), dim3(256), 0, stream>>>(hist1, 2 * NBINS);

  // 2) bf16 hi/lo splits (+ transposed expert for the column gram)
  split_kernel<<<dim3((int)(ND / 256)), dim3(256), 0, stream>>>(state, Sh, Sl, (int)ND);
  split_t_kernel<<<dim3((int)(ND / 256)), dim3(256), 0, stream>>>(expert, Eh, El, Eth, Etl, N, D);

  // 3) norms
  rownorm_kernel<<<dim3(N), dim3(256), 0, stream>>>(state, D, s2);
  rownorm_kernel<<<dim3(N), dim3(256), 0, stream>>>(expert, D, e2);
  colnorm_kernel<<<dim3(D), dim3(256), 0, stream>>>(expert, N, D, c2);

  // 4) histograms of D_es (4096x4096, K=512) and column-gram D (512x512, K=4096)
  dist_hist_kernel<<<dim3(N / 16, N / 256), dim3(256), 0, stream>>>(
      Eh, El, e2, Sh, Sl, s2, D, D, 1.0f / (float)D, hist1);
  dist_hist_kernel<<<dim3(D / 16, D / 256), dim3(256), 0, stream>>>(
      Eth, Etl, c2, Eth, Etl, c2, N, N, 1.0f / (float)N, hist2);

  // 5) medians -> gamma_1, gamma_2
  median_gamma_kernel<<<dim3(1), dim3(32), 0, stream>>>(
      hist1, (unsigned int)((size_t)N * N), hist2, (unsigned int)(D * D), gam);

  // 6) exp sums: similarity (expert rows x state cols) and self-similarity
  expsum_kernel<<<dim3(N / 16, 4), dim3(256), 0, stream>>>(
      Eh, El, e2, Sh, Sl, s2, D, D, 1.0f / (float)D, N, gam, Pes);
  expsum_kernel<<<dim3(N / 16, 4), dim3(256), 0, stream>>>(
      Sh, Sl, s2, Sh, Sl, s2, D, D, 1.0f / (float)D, N, gam, Pss);

  // 7) out[j] = mean similarity - mean self-similarity
  finalize_kernel<<<dim3(N / 256), dim3(256), 0, stream>>>(Pes, Pss, out, N);
}
